// Classifier_60662118089200
// MI455X (gfx1250) — matrix operand order repaired and checked by reading, not compiled, not run on hardware
//
#include <hip/hip_runtime.h>
#include <hip/hip_bf16.h>

typedef __attribute__((ext_vector_type(16))) _Float16 v16h;
typedef __attribute__((ext_vector_type(8)))  _Float16 v8h;
typedef __attribute__((ext_vector_type(8)))  float    v8f;

#define NN 50000
#define EE 800000
#define DD 128
#define NG 256
#define NC 10

// ---------------- degree / norm ----------------
__global__ void k_init_one(float* p, int n) {
    int i = blockIdx.x * blockDim.x + threadIdx.x;
    if (i < n) p[i] = 1.0f;   // self-loop contributes 1 to every node's degree
}

__global__ void k_scatter_deg(const int* __restrict__ dst, float* __restrict__ deg, int e) {
    int i = blockIdx.x * blockDim.x + threadIdx.x;
    if (i < e) atomicAdd(&deg[dst[i]], 1.0f);
}

__global__ void k_rsqrt_inplace(float* p, int n) {
    int i = blockIdx.x * blockDim.x + threadIdx.x;
    if (i < n) { float d = p[i]; p[i] = (d > 0.0f) ? rsqrtf(d) : 0.0f; }
}

// ---------------- f32 -> f16 conversion (4 elems/thread) ----------------
__global__ void k_cvt_f16(const float* __restrict__ in, _Float16* __restrict__ out, int n4) {
    int i = blockIdx.x * blockDim.x + threadIdx.x;
    if (i < n4) {
        float4 v = ((const float4*)in)[i];
        _Float16* o = out + (size_t)i * 4;
        o[0] = (_Float16)v.x; o[1] = (_Float16)v.y;
        o[2] = (_Float16)v.z; o[3] = (_Float16)v.w;
    }
}

// bias + relu + convert (elementwise over nrows*128)
__global__ void k_bias_relu_cvt(const float* __restrict__ agg, const float* __restrict__ bias,
                                _Float16* __restrict__ out, int total) {
    int i = blockIdx.x * blockDim.x + threadIdx.x;
    if (i < total) {
        float v = agg[i] + bias[i & (DD - 1)];
        out[i] = (_Float16)(v > 0.0f ? v : 0.0f);
    }
}

__global__ void k_zero(float* p, int n) {
    int i = blockIdx.x * blockDim.x + threadIdx.x;
    if (i < n) p[i] = 0.0f;
}

// ---------------- WMMA GEMM: Out[N,128] = A_f16[N,128] @ W_f32[128,128] ----------------
// Block = 256 threads = 8 waves. Each wave: one 16-row strip x all 8 column tiles.
// W is pre-swizzled into LDS in the exact 16-bit B-fragment layout:
//   lane L holds column n = ct*16 + (L&15); element e holds K = kb*32 + (e&7) + (e>=8 ? 16 : 0) + (L>=16 ? 8 : 0)
// Per k-block: preload all 8 B fragments into distinct registers (one LDS clause,
// one wait), then issue 8 back-to-back WMMAs — avoids per-WMMA s_wait_dscnt stalls.
__global__ __launch_bounds__(256) void k_gemm_wmma(const _Float16* __restrict__ A,
                                                   const float* __restrict__ W,
                                                   float* __restrict__ Out, int nrows) {
    __shared__ _Float16 Bfrag[8][4][32][16];   // [ct][kb][lane][elem] = 32 KB

    int t = threadIdx.x;
    for (int idx = t; idx < 8 * 4 * 32 * 16; idx += 256) {
        int e    = idx & 15;
        int lane = (idx >> 4) & 31;
        int kb   = (idx >> 9) & 3;
        int ct   = idx >> 11;
        int k = kb * 32 + (e & 7) + ((e & 8) ? 16 : 0) + ((lane & 16) ? 8 : 0);
        int n = ct * 16 + (lane & 15);
        Bfrag[ct][kb][lane][e] = (_Float16)W[k * DD + n];
    }
    __syncthreads();

    int wave = t >> 5;
    int lane = t & 31;
    int row0 = (blockIdx.x * 8 + wave) * 16;
    if (row0 >= nrows) return;                 // N % 16 == 0, so tiles are always full

    int m       = lane & 15;
    int halfsel = (lane >> 4) & 1;             // lane half selects K sub-range
    const _Float16* arow = A + (size_t)(row0 + m) * DD;

    v8f acc[8] = {};

    for (int kb = 0; kb < 4; ++kb) {
        // A fragment: elems 0-7 -> K = kb*32 + halfsel*8 + {0..7}
        //             elems 8-15 -> K = kb*32 + 16 + halfsel*8 + {0..7}
        const _Float16* ap = arow + kb * 32 + halfsel * 8;
        v8h alo = *(const v8h*)(ap);
        v8h ahi = *(const v8h*)(ap + 16);
        v16h a;
#pragma unroll
        for (int i = 0; i < 8; ++i) { a[i] = alo[i]; a[8 + i] = ahi[i]; }

        // preload all 8 B fragments for this k-block into distinct registers
        v16h bfr[8];
#pragma unroll
        for (int ct = 0; ct < 8; ++ct) {
            const _Float16* bp = &Bfrag[ct][kb][lane][0];
            v8h blo = *(const v8h*)(bp);
            v8h bhi = *(const v8h*)(bp + 8);
#pragma unroll
            for (int i = 0; i < 8; ++i) { bfr[ct][i] = blo[i]; bfr[ct][8 + i] = bhi[i]; }
        }

        // 8 back-to-back WMMAs, no intervening LDS waits
#pragma unroll
        for (int ct = 0; ct < 8; ++ct) {
            acc[ct] = __builtin_amdgcn_wmma_f32_16x16x32_f16(
                false, a, false, bfr[ct], (short)0, acc[ct], false, false);
        }
    }

    // D layout: lane col = lane&15, VGPR r -> row r (+8 for upper lane half)
    int rbase = row0 + ((lane & 16) ? 8 : 0);
#pragma unroll
    for (int ct = 0; ct < 8; ++ct) {
        float* op = Out + (size_t)rbase * DD + ct * 16 + m;
#pragma unroll
        for (int r = 0; r < 8; ++r) op[(size_t)r * DD] = acc[ct][r];
    }
}

// ---------------- edge scatter: agg[dst] += h[src] * dinv[src]*dinv[dst] ----------------
__global__ void k_scatter_edges(const int* __restrict__ src, const int* __restrict__ dst,
                                const float* __restrict__ dinv, const float* __restrict__ h,
                                float* __restrict__ agg, int e) {
    int t = blockIdx.x * blockDim.x + threadIdx.x;
    int ed = t >> 2;
    if (ed >= e) return;
    int c = (t & 3) * 32;
    int s = src[ed], d = dst[ed];
    float w = dinv[s] * dinv[d];
    const float* hp = h + (size_t)s * DD + c;
    float*       ap = agg + (size_t)d * DD + c;
#pragma unroll
    for (int j = 0; j < 32; j += 4) {
        float4 v = *(const float4*)(hp + j);
        atomicAdd(ap + j + 0, v.x * w);
        atomicAdd(ap + j + 1, v.y * w);
        atomicAdd(ap + j + 2, v.z * w);
        atomicAdd(ap + j + 3, v.w * w);
    }
}

// self-loops: agg[n] += h[n] * dinv[n]^2  (separate kernel -> no race, plain RMW)
__global__ void k_self_loops(const float* __restrict__ dinv, const float* __restrict__ h,
                             float* __restrict__ agg, int n) {
    int t = blockIdx.x * blockDim.x + threadIdx.x;
    int node = t >> 2;
    if (node >= n) return;
    int c = (t & 3) * 32;
    float w = dinv[node] * dinv[node];
    size_t base = (size_t)node * DD + c;
#pragma unroll
    for (int j = 0; j < 32; ++j) agg[base + j] += h[base + j] * w;
}

// ---------------- pooling + classifier ----------------
__global__ void k_pool(const float* __restrict__ agg, const float* __restrict__ bias,
                       const int* __restrict__ batch, float* __restrict__ sums,
                       float* __restrict__ counts, int total) {
    int i = blockIdx.x * blockDim.x + threadIdx.x;   // i = node*128 + j
    if (i >= total) return;
    int node = i >> 7;
    int j = i & (DD - 1);
    float v = agg[i] + bias[j];
    v = v > 0.0f ? v : 0.0f;
    int g = batch[node];
    atomicAdd(&sums[(size_t)g * DD + j], v);
    if (j == 0) atomicAdd(&counts[g], 1.0f);
}

__global__ void k_classify(const float* __restrict__ sums, const float* __restrict__ counts,
                           const float* __restrict__ Wc, const float* __restrict__ bc,
                           float* __restrict__ out) {
    int t = blockIdx.x * blockDim.x + threadIdx.x;
    if (t >= NG * NC) return;
    int g = t / NC, c = t % NC;
    float cnt = counts[g];
    float inv = 1.0f / (cnt > 1.0f ? cnt : 1.0f);
    float acc = bc[c];
    const float* sg = sums + (size_t)g * DD;
    for (int k = 0; k < DD; ++k) acc += sg[k] * inv * Wc[k * NC + c];
    out[t] = acc;
}

// ---------------- launcher ----------------
extern "C" void kernel_launch(void* const* d_in, const int* in_sizes, int n_in,
                              void* d_out, int out_size, void* d_ws, size_t ws_size,
                              hipStream_t stream) {
    (void)in_sizes; (void)n_in; (void)out_size; (void)ws_size;

    const float* x   = (const float*)d_in[0];
    const int*   ei  = (const int*)d_in[1];      // [2, E]: row0 = src, row1 = dst
    const int*   bat = (const int*)d_in[2];
    const float* W1  = (const float*)d_in[3];
    const float* b1  = (const float*)d_in[4];
    const float* W2  = (const float*)d_in[5];
    const float* b2  = (const float*)d_in[6];
    const float* Wc  = (const float*)d_in[7];
    const float* bc  = (const float*)d_in[8];
    float* out = (float*)d_out;

    const int* src = ei;
    const int* dst = ei + EE;

    // workspace carve-up (~64.3 MB)
    float*    dinv = (float*)d_ws;                              // N
    _Float16* Ah   = (_Float16*)(dinv + NN);                    // N*128 f16
    float*    H    = (float*)(Ah + (size_t)NN * DD);            // N*128 f32
    float*    AGG  = H + (size_t)NN * DD;                       // N*128 f32
    float*    sums = AGG + (size_t)NN * DD;                     // 256*128
    float*    cnts = sums + (size_t)NG * DD;                    // 256

    const int total = NN * DD;          // 6.4M
    const int B = 256;
    dim3 blk(B);

    // degrees -> dinv
    k_init_one<<<dim3((NN + B - 1) / B), blk, 0, stream>>>(dinv, NN);
    k_scatter_deg<<<dim3((EE + B - 1) / B), blk, 0, stream>>>(dst, dinv, EE);
    k_rsqrt_inplace<<<dim3((NN + B - 1) / B), blk, 0, stream>>>(dinv, NN);

    // ---- conv1 ----
    k_cvt_f16<<<dim3((total / 4 + B - 1) / B), blk, 0, stream>>>(x, Ah, total / 4);
    k_gemm_wmma<<<dim3((NN + 127) / 128), blk, 0, stream>>>(Ah, W1, H, NN);
    k_zero<<<dim3((total + B - 1) / B), blk, 0, stream>>>(AGG, total);
    k_scatter_edges<<<dim3((EE * 4 + B - 1) / B), blk, 0, stream>>>(src, dst, dinv, H, AGG, EE);
    k_self_loops<<<dim3((NN * 4 + B - 1) / B), blk, 0, stream>>>(dinv, H, AGG, NN);

    // ---- conv2 ----
    k_bias_relu_cvt<<<dim3((total + B - 1) / B), blk, 0, stream>>>(AGG, b1, Ah, total);
    k_gemm_wmma<<<dim3((NN + 127) / 128), blk, 0, stream>>>(Ah, W2, H, NN);
    k_zero<<<dim3((total + B - 1) / B), blk, 0, stream>>>(AGG, total);
    k_scatter_edges<<<dim3((EE * 4 + B - 1) / B), blk, 0, stream>>>(src, dst, dinv, H, AGG, EE);
    k_self_loops<<<dim3((NN * 4 + B - 1) / B), blk, 0, stream>>>(dinv, H, AGG, NN);

    // ---- pool + classify ----
    k_zero<<<dim3((NG * DD + NG + B - 1) / B), blk, 0, stream>>>(sums, NG * DD + NG); // sums+counts contiguous
    k_pool<<<dim3((total + B - 1) / B), blk, 0, stream>>>(AGG, b2, bat, sums, cnts, total);
    k_classify<<<dim3((NG * NC + B - 1) / B), blk, 0, stream>>>(sums, cnts, Wc, bc, out);
}
